// GMM_score_7112465842631
// MI455X (gfx1250) — matrix operand order repaired and checked by reading, not compiled, not run on hardware
//
#include <hip/hip_runtime.h>
#include <hip/hip_bf16.h>
#include <cstdint>
#include <cstddef>

// ---------------------------------------------------------------------------
// GMM score (VP-SDE marginals) == flash-attention with per-row scale/bias.
//   scores[n,m] = a_n * <x_n, y_m> - b_n * ||y_m||^2   (softmax-equivalent)
//   out = (softmax(scores) @ Y - x) * inv_var_n
// bf16 WMMA for both GEMMs; split-M online softmax; async global->LDS
// double-buffered staging of pre-swizzled B fragments (ASYNCcnt pipeline).
// ---------------------------------------------------------------------------

typedef __attribute__((ext_vector_type(16))) __bf16 v16bf;
typedef __attribute__((ext_vector_type(8)))  float  v8f;

union Frag16 {
  v16bf v;
  unsigned short u[16];
  uint4 q[2];
};

__device__ __forceinline__ unsigned short f2bf(float f) {
  unsigned u = __float_as_uint(f);
  unsigned r = u + 0x7FFFu + ((u >> 16) & 1u);   // round-to-nearest-even
  return (unsigned short)(r >> 16);
}

constexpr int N = 4096, M = 16384, D = 128;
constexpr int SPLIT = 4;                 // split-softmax partitions over M
constexpr float BETA_MIN = 0.1f, BETA_MAX = 20.0f;

__device__ __forceinline__ void marginals(float t, float& mean, float& inv_var) {
  float log_mean = -0.25f * t * t * (BETA_MAX - BETA_MIN) - 0.5f * t * BETA_MIN;
  mean = __expf(log_mean);
  float var = fmaxf(1.0f - mean * mean, 1e-12f);
  inv_var = 1.0f / var;
}

// ---------------------------------------------------------------------------
// Precompute: rewrite train_data into WMMA B-fragment register images (bf16).
// Bqk: B for QK^T  (K = d).  lane L, dword j holds y[16*mt + L%16][32c + 16*(L/16) + 2j (+1)]
// Bpv: B for P@V   (K = m).  lane L, dword j holds y[32*mk + 16*(L/16) + 2j (+1)][16*dt + L%16]
// Per 64-m slab, each region is one contiguous 16 KB block (m0*256 bytes in).
// ---------------------------------------------------------------------------
__global__ void prep_bqk(const float* __restrict__ train, unsigned* __restrict__ bqk) {
  int idx = blockIdx.x * blockDim.x + threadIdx.x;   // dword index
  int j  = idx & 7;
  int L  = (idx >> 3) & 31;
  int c  = (idx >> 8) & 3;
  int mt = idx >> 10;
  int m  = mt * 16 + (L & 15);
  int d  = 32 * c + 16 * (L >> 4) + 2 * j;
  const float* r = train + m * D + d;
  unsigned lo = f2bf(r[0]), hi = f2bf(r[1]);
  bqk[idx] = lo | (hi << 16);
}

__global__ void prep_bpv(const float* __restrict__ train, unsigned* __restrict__ bpv) {
  int idx = blockIdx.x * blockDim.x + threadIdx.x;   // dword index
  int j  = idx & 7;
  int L  = (idx >> 3) & 31;
  int dt = (idx >> 8) & 7;
  int mk = idx >> 11;
  int d  = 16 * dt + (L & 15);
  int m0 = 32 * mk + 16 * (L >> 4) + 2 * j;
  unsigned lo = f2bf(train[m0 * D + d]);
  unsigned hi = f2bf(train[(m0 + 1) * D + d]);
  bpv[idx] = lo | (hi << 16);
}

__global__ void prep_y2(const float* __restrict__ train, float* __restrict__ y2) {
  int m = blockIdx.x * blockDim.x + threadIdx.x;
  const float* r = train + m * D;
  float s = 0.f;
#pragma unroll 8
  for (int d = 0; d < D; ++d) s += r[d] * r[d];
  y2[m] = s;
}

// ---------------------------------------------------------------------------
// Main fused kernel: one wave = 16 rows of x; blockIdx.y = M-partition.
// Async double-buffered LDS staging; 32 bf16 WMMAs per 64-m iteration/wave.
// ---------------------------------------------------------------------------
__global__ __launch_bounds__(128) void attn_partial(
    const float* __restrict__ x, const float* __restrict__ t,
    const unsigned* __restrict__ bqk, const unsigned* __restrict__ bpv,
    const float* __restrict__ y2g, float* __restrict__ opart,
    float* __restrict__ mpart, float* __restrict__ spart) {
  __shared__ __align__(16) char stage[2][32768];           // [16K Bqk | 16K Bpv]
  __shared__ __align__(16) unsigned short ldsP[4][16][64]; // per-wave P tile (bf16)

  const int tid  = threadIdx.x;
  const int wave = tid >> 5;
  const int lane = tid & 31;
  const int hi16 = lane >> 4;        // lane group (0: lanes 0-15, 1: lanes 16-31)
  const int lm   = lane & 15;
  const int n0   = blockIdx.x * 64 + wave * 16;
  const int part = blockIdx.y;
  const int mBeg = part * (M / SPLIT);
  const int nIter = (M / SPLIT) / 64;

  const uint32_t ldsStage0 = (uint32_t)(size_t)&stage[0][0] + (uint32_t)tid * 16u;

  // Issue the async copy of one 64-m slab (32 KB) into LDS buffer `buf`.
  // Each thread moves 16 x 16B chunks; IOFFSET advances BOTH the LDS and the
  // global address (ISA 10.7), so per-thread state is just two base addresses.
  auto issueCopy = [&](int buf, int m0c) {
    uint64_t gq = (uint64_t)(size_t)bqk + (size_t)m0c * 256 + (uint32_t)tid * 16u;
    uint64_t gp = (uint64_t)(size_t)bpv + (size_t)m0c * 256 + (uint32_t)tid * 16u;
    uint32_t lq = ldsStage0 + (uint32_t)buf * 32768u;
    uint32_t lp = lq + 16384u;
#define CP(Lr, Gr, OFF) \
    asm volatile("global_load_async_to_lds_b128 %0, %1, off offset:" OFF \
                 :: "v"(Lr), "v"(Gr) : "memory")
    CP(lq, gq, "0");     CP(lq, gq, "2048");  CP(lq, gq, "4096");  CP(lq, gq, "6144");
    CP(lq, gq, "8192");  CP(lq, gq, "10240"); CP(lq, gq, "12288"); CP(lq, gq, "14336");
    CP(lp, gp, "0");     CP(lp, gp, "2048");  CP(lp, gp, "4096");  CP(lp, gp, "6144");
    CP(lp, gp, "8192");  CP(lp, gp, "10240"); CP(lp, gp, "12288"); CP(lp, gp, "14336");
#undef CP
  };

  // Per-slot row constants: C-matrix slot i <-> row n0 + i + 8*hi16
  float aCoef[8], bCoef[8], rmax[8], lsum[8];
#pragma unroll
  for (int i = 0; i < 8; ++i) {
    int n = n0 + i + 8 * hi16;
    float mean, iv; marginals(t[n], mean, iv);
    aCoef[i] = mean * iv;
    bCoef[i] = 0.5f * mean * mean * iv;
    rmax[i]  = -__builtin_inff();
    lsum[i]  = 0.f;
  }

  // x fragments (A-matrix, 16x32 bf16, k = 16*hi16 + s): 4 K-chunks over D=128
  Frag16 xf[4];
  {
    const float* xr = x + (n0 + lm) * D;
#pragma unroll
    for (int c = 0; c < 4; ++c) {
      int da = 32 * c + 16 * hi16;
#pragma unroll
      for (int j = 0; j < 8; ++j) xf[c].u[j]     = f2bf(xr[da + j]);
#pragma unroll
      for (int j = 0; j < 8; ++j) xf[c].u[8 + j] = f2bf(xr[da + 8 + j]);
    }
  }

  v8f O[8];
#pragma unroll
  for (int dt = 0; dt < 8; ++dt)
#pragma unroll
    for (int e = 0; e < 8; ++e) O[dt][e] = 0.f;

  issueCopy(0, mBeg);                       // prologue: slab 0 -> buffer 0

  for (int it = 0; it < nIter; ++it) {
    const int m0 = mBeg + (it << 6);
    // Prefetch next slab into the other buffer (clamped re-issue on last iter
    // keeps the ASYNCcnt arithmetic uniform; the target buffer is idle).
    const int mNext = (it + 1 < nIter) ? (m0 + 64) : m0;
    issueCopy((it + 1) & 1, mNext);
    asm volatile("s_wait_asynccnt 0x10" ::: "memory");  // current slab landed
    __syncthreads();                                    // ...for all waves

    const uint4* sq = (const uint4*)(&stage[it & 1][0]);      // Bqk frags
    const uint4* sv = (const uint4*)(&stage[it & 1][16384]);  // Bpv frags

    // ---- S = x @ Y^T tile: 4 column-tiles x 4 K-chunks = 16 WMMA ----
    v8f S[4];
#pragma unroll
    for (int tt = 0; tt < 4; ++tt) {
      v8f acc;
#pragma unroll
      for (int e = 0; e < 8; ++e) acc[e] = 0.f;
#pragma unroll
      for (int c = 0; c < 4; ++c) {
        Frag16 bf_;
        int base = ((tt * 4 + c) * 32 + lane) * 2;   // uint4 index
        bf_.q[0] = sq[base];
        bf_.q[1] = sq[base + 1];
        acc = __builtin_amdgcn_wmma_f32_16x16x32_bf16(false, xf[c].v, false, bf_.v,
                                                      (short)0, acc, false, false);
      }
      S[tt] = acc;
    }

    // ---- online softmax over this 64-column slab ----
    float nm[8];
#pragma unroll
    for (int i = 0; i < 8; ++i) nm[i] = rmax[i];
    float y2v[4];
#pragma unroll
    for (int tt = 0; tt < 4; ++tt) y2v[tt] = y2g[m0 + tt * 16 + lm];
#pragma unroll
    for (int tt = 0; tt < 4; ++tt)
#pragma unroll
      for (int i = 0; i < 8; ++i) {
        float s = aCoef[i] * S[tt][i] - bCoef[i] * y2v[tt];
        S[tt][i] = s;
        nm[i] = fmaxf(nm[i], s);
      }
    // row-max across the 16 lanes sharing each row (half-wave groups)
#pragma unroll
    for (int msk = 1; msk < 16; msk <<= 1)
#pragma unroll
      for (int i = 0; i < 8; ++i)
        nm[i] = fmaxf(nm[i], __shfl_xor(nm[i], msk, 16));
#pragma unroll
    for (int i = 0; i < 8; ++i) {
      float sc = __expf(rmax[i] - nm[i]);
      rmax[i] = nm[i];
      lsum[i] *= sc;
#pragma unroll
      for (int dt = 0; dt < 8; ++dt) O[dt][i] *= sc;
    }
    // P = exp(S - rowmax): per-lane partial sums, spill tile to LDS (bf16)
#pragma unroll
    for (int tt = 0; tt < 4; ++tt)
#pragma unroll
      for (int i = 0; i < 8; ++i) {
        float p = __expf(S[tt][i] - nm[i]);
        lsum[i] += p;
        ldsP[wave][i + 8 * hi16][tt * 16 + lm] = f2bf(p);
      }
    asm volatile("s_wait_dscnt 0x0" ::: "memory");   // wave-private LDS RAW

    // ---- O += P @ V: reload P in A-layout, 8 d-tiles x 2 K-chunks = 16 WMMA ----
    Frag16 pf[2];
#pragma unroll
    for (int k = 0; k < 2; ++k) {
      const uint4* p0 = (const uint4*)&ldsP[wave][lm][32 * k + 16 * hi16];
      const uint4* p1 = (const uint4*)&ldsP[wave][lm][32 * k + 16 * hi16 + 8];
      pf[k].q[0] = p0[0];
      pf[k].q[1] = p1[0];
    }
#pragma unroll
    for (int dt = 0; dt < 8; ++dt)
#pragma unroll
      for (int k = 0; k < 2; ++k) {
        Frag16 vf;
        int base = ((k * 8 + dt) * 32 + lane) * 2;   // uint4 index
        vf.q[0] = sv[base];
        vf.q[1] = sv[base + 1];
        O[dt] = __builtin_amdgcn_wmma_f32_16x16x32_bf16(false, pf[k].v, false, vf.v,
                                                        (short)0, O[dt], false, false);
      }

    __syncthreads();   // all waves done with this buffer before it is re-filled
  }

  // finalize: sum partial softmax denominators across the 16 lanes per row
#pragma unroll
  for (int msk = 1; msk < 16; msk <<= 1)
#pragma unroll
    for (int i = 0; i < 8; ++i) lsum[i] += __shfl_xor(lsum[i], msk, 16);

#pragma unroll
  for (int dt = 0; dt < 8; ++dt)
#pragma unroll
    for (int i = 0; i < 8; ++i) {
      int n = n0 + i + 8 * hi16;
      opart[((size_t)part * N + n) * D + dt * 16 + lm] = O[dt][i];
    }
  if (lm == 0) {
#pragma unroll
    for (int i = 0; i < 8; ++i) {
      int n = n0 + i + 8 * hi16;
      mpart[part * N + n] = rmax[i];
      spart[part * N + n] = lsum[i];
    }
  }
}

// ---------------------------------------------------------------------------
// Merge the SPLIT partial softmaxes (log-sum-exp combine) + epilogue.
// ---------------------------------------------------------------------------
__global__ void merge_out(const float* __restrict__ x, const float* __restrict__ t,
                          const float* __restrict__ opart, const float* __restrict__ mpart,
                          const float* __restrict__ spart, float* __restrict__ out) {
  int idx = blockIdx.x * blockDim.x + threadIdx.x;
  if (idx >= N * D) return;
  int n = idx >> 7;
  int d = idx & (D - 1);
  float gmax = -__builtin_inff();
#pragma unroll
  for (int s = 0; s < SPLIT; ++s) gmax = fmaxf(gmax, mpart[s * N + n]);
  float denom = 0.f, acc = 0.f;
#pragma unroll
  for (int s = 0; s < SPLIT; ++s) {
    float e = __expf(mpart[s * N + n] - gmax);
    denom += spart[s * N + n] * e;
    acc   += opart[((size_t)s * N + n) * D + d] * e;
  }
  float ev = acc / denom;
  if (ev != ev) ev = 0.f;                 // reference: NaN -> 0
  float mean, iv; marginals(t[n], mean, iv);
  out[idx] = (ev - x[idx]) * iv;
}

// ---------------------------------------------------------------------------
extern "C" void kernel_launch(void* const* d_in, const int* in_sizes, int n_in,
                              void* d_out, int out_size, void* d_ws, size_t ws_size,
                              hipStream_t stream) {
  (void)in_sizes; (void)n_in; (void)out_size; (void)ws_size;
  const float* x     = (const float*)d_in[0];
  const float* t     = (const float*)d_in[1];
  const float* train = (const float*)d_in[2];
  float* out = (float*)d_out;

  char* ws = (char*)d_ws;
  unsigned* bqk   = (unsigned*)(ws);                                   //  4 MB
  unsigned* bpv   = (unsigned*)(ws + ((size_t)4  << 20));              //  4 MB
  float*    y2    = (float*)   (ws + ((size_t)8  << 20));              // 64 KB
  float*    opart = (float*)   (ws + ((size_t)8  << 20) + (64u << 10));//  8 MB
  float*    mpart = (float*)   (ws + ((size_t)16 << 20) + (64u << 10));// 64 KB
  float*    spart = (float*)   (ws + ((size_t)16 << 20) + (128u << 10));//64 KB

  prep_bqk<<<(M / 16) * 4 * 32 * 8 / 256, 256, 0, stream>>>(train, bqk);
  prep_bpv<<<(M / 32) * 8 * 32 * 8 / 256, 256, 0, stream>>>(train, bpv);
  prep_y2 <<<M / 256, 256, 0, stream>>>(train, y2);

  dim3 grid(N / 64, SPLIT);
  attn_partial<<<grid, 128, 0, stream>>>(x, t, bqk, bpv, y2, opart, mpart, spart);

  merge_out<<<(N * D) / 256, 256, 0, stream>>>(x, t, opart, mpart, spart, out);
}
